// GCN_46840913330200
// MI455X (gfx1250) — matrix operand order repaired and checked by reading, not compiled, not run on hardware
//
#include <hip/hip_runtime.h>
#include <hip/hip_bf16.h>

// ---------------------------------------------------------------------------
// GCN on MI455X (gfx1250): WMMA f16->f32 GEMMs, TDM-staged weights,
// L2-resident edge aggregation with f32 atomics.
// ---------------------------------------------------------------------------

typedef __attribute__((ext_vector_type(16))) _Float16 v16h;
typedef __attribute__((ext_vector_type(8)))  float    v8f;
typedef __attribute__((ext_vector_type(4)))  _Float16 v4h;
typedef __attribute__((ext_vector_type(4)))  unsigned int u32x4;
typedef __attribute__((ext_vector_type(8)))  int i32x8;
typedef __attribute__((ext_vector_type(4)))  int i32x4;

#define H 128          // hidden dim (fixed by reference)
#define FIN 6          // input feature dim
#define SBTS 136       // padded LDS row stride in halves (68 dwords: bank-conflict free)

#if defined(__HIP_DEVICE_COMPILE__) && __has_builtin(__builtin_amdgcn_tensor_load_to_lds)
#define GCN_USE_TDM 1
#else
#define GCN_USE_TDM 0
#endif

// ---------------------------------------------------------------- utilities

__global__ void deg_init_kernel(float* deg, int n) {
    int i = blockIdx.x * blockDim.x + threadIdx.x;
    if (i < n) deg[i] = 1.0f;                       // self-loop contributes 1
}

__global__ void deg_edges_kernel(const int* dst, float* deg, int e) {
    int i = blockIdx.x * blockDim.x + threadIdx.x;
    if (i < e) atomicAdd(&deg[dst[i]], 1.0f);
}

__global__ void dinv_kernel(float* deg_dinv, int n) {
    int i = blockIdx.x * blockDim.x + threadIdx.x;
    if (i < n) {
        float d = deg_dinv[i];
        deg_dinv[i] = (d > 0.0f) ? __frsqrt_rn(d) : 0.0f;
    }
}

// f32 row-major [k][n] -> f16 transposed [n][k]  (so B K-pairs are contiguous)
__global__ void wcvt_kernel(const float* __restrict__ w, _Float16* __restrict__ wt) {
    int i = blockIdx.x * blockDim.x + threadIdx.x;   // i = k*H + n
    if (i < H * H) {
        int k = i >> 7, n = i & (H - 1);
        wt[n * H + k] = (_Float16)w[i];
    }
}

// Layer 0: t[n][f] = sum_{k<6} x[n][k] * W0[k][f]   (K=6: plain VALU)
__global__ void lin0_kernel(const float* __restrict__ x, const float* __restrict__ w0,
                            float* __restrict__ t, int n) {
    int i = blockIdx.x * blockDim.x + threadIdx.x;
    if (i >= n * H) return;
    int node = i >> 7, f = i & (H - 1);
    const float* xr = x + node * FIN;
    float acc = 0.0f;
#pragma unroll
    for (int k = 0; k < FIN; ++k) acc = fmaf(xr[k], w0[k * H + f], acc);
    t[i] = acc;
}

// agg = t * dinv^2   (self-loop term; also serves as the zero-init of agg)
__global__ void agg_init_kernel(const float4* __restrict__ t, const float* __restrict__ dinv,
                                float4* __restrict__ agg, int n) {
    int i = blockIdx.x * blockDim.x + threadIdx.x;   // over n*32 float4s
    if (i >= n * (H / 4)) return;
    int node = i >> 5;
    float s = dinv[node]; s = s * s;
    float4 v = t[i];
    v.x *= s; v.y *= s; v.z *= s; v.w *= s;
    agg[i] = v;
}

// one wave per edge: agg[dst] += t[src] * (dinv[src]*dinv[dst])
__global__ void agg_edges_kernel(const int* __restrict__ src, const int* __restrict__ dst,
                                 const float* __restrict__ dinv,
                                 const float* __restrict__ t, float* __restrict__ agg, int e) {
    int wave = blockIdx.x * (blockDim.x >> 5) + (threadIdx.x >> 5);
    if (wave >= e) return;
    int lane = threadIdx.x & 31;
    int s = src[wave], d = dst[wave];
    float w = dinv[s] * dinv[d];
    const float4* ts = (const float4*)(t + (size_t)s * H);
    float4 v = ts[lane];
    float* ad = agg + (size_t)d * H + lane * 4;
    atomicAdd(ad + 0, v.x * w);
    atomicAdd(ad + 1, v.y * w);
    atomicAdd(ad + 2, v.z * w);
    atomicAdd(ad + 3, v.w * w);
}

// h16 = f16( relu(agg + b) ), 4 features per thread
__global__ void finalize_relu_kernel(const float4* __restrict__ agg, const float4* __restrict__ b,
                                     v4h* __restrict__ h16, int n) {
    int i = blockIdx.x * blockDim.x + threadIdx.x;   // over n*32 float4s
    if (i >= n * (H / 4)) return;
    float4 v = agg[i];
    float4 bb = b[i & 31];
    v.x += bb.x; v.y += bb.y; v.z += bb.z; v.w += bb.w;
    v4h o;
    o[0] = (_Float16)(v.x > 0.0f ? v.x : 0.0f);
    o[1] = (_Float16)(v.y > 0.0f ? v.y : 0.0f);
    o[2] = (_Float16)(v.z > 0.0f ? v.z : 0.0f);
    o[3] = (_Float16)(v.w > 0.0f ? v.w : 0.0f);
    h16[i] = o;
}

// out[n] = Rb2 + sum_f h16[n][f]*Rw2[f]
__global__ void final_dot_kernel(const _Float16* __restrict__ h16, const float* __restrict__ rw2,
                                 const float* __restrict__ rb2, float* __restrict__ out, int n) {
    int i = blockIdx.x * blockDim.x + threadIdx.x;
    if (i >= n) return;
    const unsigned int* hp = (const unsigned int*)(h16 + (size_t)i * H);
    float acc = 0.0f;
#pragma unroll 8
    for (int j = 0; j < H / 2; ++j) {
        union { unsigned int u; _Float16 h[2]; } t;
        t.u = hp[j];
        acc = fmaf((float)t.h[0], rw2[2 * j], acc);
        acc = fmaf((float)t.h[1], rw2[2 * j + 1], acc);
    }
    out[i] = acc + rb2[0];
}

// ------------------------------------------------------------- WMMA GEMM
// C[16 x 128] slab per block: A(16xK f16) @ B(KxN f16), K=N=128.
// 8 waves, wave w owns columns [16w,16w+16). A block + full B^T staged in LDS
// (B^T via the Tensor Data Mover when available, with HW row padding).
// MODE 0: store f32 (conv pre-aggregation).  MODE 1: +bias, leaky-relu, f16.

template <int MODE>
__global__ void gemm_wmma_kernel(const _Float16* __restrict__ a16,   // [n x 128]
                                 const _Float16* __restrict__ bt16,  // [128 x 128] n-major
                                 const float* __restrict__ bias,     // [128] or null
                                 float* __restrict__ out32,          // MODE 0
                                 _Float16* __restrict__ out16,       // MODE 1
                                 int n) {
    __shared__ _Float16 sBT[H * SBTS];   // 34 KB: B^T, padded rows (k contiguous per column)
    __shared__ _Float16 sA[16 * SBTS];   // 4.25 KB: 16-row A block, padded rows

    const int tid  = threadIdx.x;
    const int lane = tid & 31;
    const int wv   = tid >> 5;           // 0..7
    const int rowBase = blockIdx.x * 16;

#if GCN_USE_TDM
    // One TDM descriptor per block: 1-D tile of 16384 f16 (32 KB) with LDS
    // padding of 4 dwords every 64 dwords -> padded SBTS=136-half rows.
    if (tid == 0) {
        unsigned long long ga = (unsigned long long)(const void*)bt16;
        unsigned int ldsoff = (unsigned int)(unsigned long long)(void*)&sBT[0];
        u32x4 g0;
        g0[0] = 1u;                                           // count=1 (user D#)
        g0[1] = ldsoff;                                       // lds_addr
        g0[2] = (unsigned int)ga;                             // global_addr[31:0]
        g0[3] = (unsigned int)((ga >> 32) & 0x01FFFFFFull)    // global_addr[56:32]
              | 0x80000000u;                                  // type=2 ("image")
        i32x8 g1;
        g1[0] = 0x07510000;      // data_size=2B | pad_enable | pad_interval=64dw | pad_amount=4dw
        g1[1] = (int)(16384u << 16);   // tensor_dim0 = 16384 elements
        g1[2] = (int)(1u << 16);       // tensor_dim1 = 1
        g1[3] = (int)(16384u << 16);   // tile_dim0 = 16384
        g1[4] = 0;                     // tile_dim1/2 unused
        g1[5] = 16384;                 // tensor_dim0_stride
        g1[6] = 0;
        g1[7] = 0;
        i32x4 gz; gz[0] = 0; gz[1] = 0; gz[2] = 0; gz[3] = 0;
#if __clang_major__ >= 23
        i32x8 gz8; gz8[0]=0; gz8[1]=0; gz8[2]=0; gz8[3]=0; gz8[4]=0; gz8[5]=0; gz8[6]=0; gz8[7]=0;
        __builtin_amdgcn_tensor_load_to_lds(g0, g1, gz, gz, gz8, 0);
#else
        __builtin_amdgcn_tensor_load_to_lds(g0, g1, gz, gz, 0);
#endif
        __builtin_amdgcn_s_wait_tensorcnt((short)0);
    }
#else
    // fallback: cooperative copy into padded layout (8192 dwords / 256 thr)
    {
        const unsigned int* gsrc = (const unsigned int*)bt16;
        unsigned int* ldst = (unsigned int*)sBT;
#pragma unroll
        for (int j = 0; j < 32; ++j) {
            int d = tid + j * 256;
            ldst[(d >> 6) * (SBTS / 2) + (d & 63)] = gsrc[d];
        }
    }
#endif
    // stage A block into padded layout: 1024 dwords / 256 threads = 4 each
    {
        unsigned int* ldst = (unsigned int*)sA;
        const unsigned int* gsrc = (const unsigned int*)(a16 + (size_t)rowBase * H);
#pragma unroll
        for (int j = 0; j < 4; ++j) {
            int d = tid + j * 256;               // dword index in 16x128 block
            int row = rowBase + (d >> 6);        // 64 dwords per logical row
            ldst[(d >> 6) * (SBTS / 2) + (d & 63)] = (row < n) ? gsrc[d] : 0u;
        }
    }
    __syncthreads();

    union { v16h v; unsigned int u[8]; } a, b;
    union { v8f v; float f[8]; } c;
#pragma unroll
    for (int r = 0; r < 8; ++r) c.f[r] = 0.0f;

    const int half = lane >> 4;          // 0 or 1
    const int mr   = lane & 15;
    const int col  = wv * 16 + mr;       // this lane's output column

#pragma unroll
    for (int kb = 0; kb < H; kb += 32) {
        // A fragment: lane<16 holds K kb+[0..7] and kb+[16..23]; lane>=16 shifts by 8
#pragma unroll
        for (int j = 0; j < 4; ++j) {
            int k0 = kb + half * 8 + 2 * j;
            int k1 = kb + 16 + half * 8 + 2 * j;
            a.u[j]     = *(const unsigned int*)&sA[mr * SBTS + k0];
            a.u[4 + j] = *(const unsigned int*)&sA[mr * SBTS + k1];
        }
        // B fragment: lane<16 holds K kb+[0..7] and kb+[16..23] of its column; lane>=16 shifts by 8
#pragma unroll
        for (int j = 0; j < 4; ++j) {
            int k0 = kb + half * 8 + 2 * j;
            int k1 = kb + 16 + half * 8 + 2 * j;
            b.u[j]     = *(const unsigned int*)&sBT[col * SBTS + k0];
            b.u[4 + j] = *(const unsigned int*)&sBT[col * SBTS + k1];
        }
        c.v = __builtin_amdgcn_wmma_f32_16x16x32_f16(
                  false, a.v, false, b.v, (short)0, c.v, false, false);
    }

    // D layout: lanes 0-15 -> M=r, lanes 16-31 -> M=8+r, N = col.
    // Uniform full-tile test keeps the common path branch-free: stores become
    // immediate-offset global_store off one address pair.
    const int rtop = rowBase + half * 8;
    if (MODE == 0) {
        float* po = out32 + (size_t)rtop * H + col;
        if (rowBase + 16 <= n) {
#pragma unroll
            for (int r = 0; r < 8; ++r) po[(size_t)r * H] = c.f[r];
        } else {
#pragma unroll
            for (int r = 0; r < 8; ++r) if (rtop + r < n) po[(size_t)r * H] = c.f[r];
        }
    } else {
        const float bc = bias[col];
        _Float16* po = out16 + (size_t)rtop * H + col;
        if (rowBase + 16 <= n) {
#pragma unroll
            for (int r = 0; r < 8; ++r) {
                float v = c.f[r] + bc;
                po[(size_t)r * H] = (_Float16)(v > 0.0f ? v : 0.01f * v);
            }
        } else {
#pragma unroll
            for (int r = 0; r < 8; ++r) {
                if (rtop + r < n) {
                    float v = c.f[r] + bc;
                    po[(size_t)r * H] = (_Float16)(v > 0.0f ? v : 0.01f * v);
                }
            }
        }
    }
}

// --------------------------------------------------------------- launcher

extern "C" void kernel_launch(void* const* d_in, const int* in_sizes, int n_in,
                              void* d_out, int out_size, void* d_ws, size_t ws_size,
                              hipStream_t stream) {
    const float* x    = (const float*)d_in[0];
    const int*   ei   = (const int*)d_in[1];
    const float* W0   = (const float*)d_in[2];
    const float* b0   = (const float*)d_in[3];
    const float* W1   = (const float*)d_in[4];
    const float* b1   = (const float*)d_in[5];
    const float* W2   = (const float*)d_in[6];
    const float* b2   = (const float*)d_in[7];
    const float* Rw0  = (const float*)d_in[8];
    const float* Rb0  = (const float*)d_in[9];
    const float* Rw1  = (const float*)d_in[10];
    const float* Rb1  = (const float*)d_in[11];
    const float* Rw2  = (const float*)d_in[12];
    const float* Rb2  = (const float*)d_in[13];
    float* out = (float*)d_out;

    const int N = in_sizes[0] / FIN;
    const int E = in_sizes[1] / 2;
    const int* src = ei;
    const int* dst = ei + E;

    // workspace carve-up (256B aligned)
    char* ws = (char*)d_ws;
    size_t off = 0;
    auto take = [&](size_t bytes) { void* p = ws + off; off += (bytes + 255) & ~size_t(255); return p; };
    float*    dinv  = (float*)take((size_t)N * 4);
    _Float16* wt1   = (_Float16*)take((size_t)H * H * 2);
    _Float16* wt2   = (_Float16*)take((size_t)H * H * 2);
    _Float16* wtR0  = (_Float16*)take((size_t)H * H * 2);
    _Float16* wtR1  = (_Float16*)take((size_t)H * H * 2);
    float*    t32   = (float*)take((size_t)N * H * 4);
    float*    agg   = (float*)take((size_t)N * H * 4);
    _Float16* h16a  = (_Float16*)take((size_t)N * H * 2);
    _Float16* h16b  = (_Float16*)take((size_t)N * H * 2);
    (void)ws_size;

    const int TB = 256;
    dim3 blkN((N + TB - 1) / TB);
    dim3 blkE((E + TB - 1) / TB);
    dim3 blkNH(((size_t)N * H + TB - 1) / TB);
    dim3 blkN4(((size_t)N * (H / 4) + TB - 1) / TB);
    dim3 blkEdgeWave((E + 7) / 8);          // 8 waves / block, one edge / wave
    dim3 blkHH((H * H + TB - 1) / TB);
    dim3 blkGemm((N + 15) / 16);

    // degrees -> dinv (in place)
    deg_init_kernel<<<blkN, TB, 0, stream>>>(dinv, N);
    deg_edges_kernel<<<blkE, TB, 0, stream>>>(dst, dinv, E);
    dinv_kernel<<<blkN, TB, 0, stream>>>(dinv, N);

    // weight conversion (f32 -> transposed f16)
    wcvt_kernel<<<blkHH, TB, 0, stream>>>(W1, wt1);
    wcvt_kernel<<<blkHH, TB, 0, stream>>>(W2, wt2);
    wcvt_kernel<<<blkHH, TB, 0, stream>>>(Rw0, wtR0);
    wcvt_kernel<<<blkHH, TB, 0, stream>>>(Rw1, wtR1);

    // ---- conv layer 0 (K=6 linear, then aggregate) ----
    lin0_kernel<<<blkNH, TB, 0, stream>>>(x, W0, t32, N);
    agg_init_kernel<<<blkN4, TB, 0, stream>>>((const float4*)t32, dinv, (float4*)agg, N);
    agg_edges_kernel<<<blkEdgeWave, TB, 0, stream>>>(src, dst, dinv, t32, agg, E);
    finalize_relu_kernel<<<blkN4, TB, 0, stream>>>((const float4*)agg, (const float4*)b0, (v4h*)h16a, N);

    // ---- conv layer 1 ----
    gemm_wmma_kernel<0><<<blkGemm, TB, 0, stream>>>(h16a, wt1, nullptr, t32, nullptr, N);
    agg_init_kernel<<<blkN4, TB, 0, stream>>>((const float4*)t32, dinv, (float4*)agg, N);
    agg_edges_kernel<<<blkEdgeWave, TB, 0, stream>>>(src, dst, dinv, t32, agg, E);
    finalize_relu_kernel<<<blkN4, TB, 0, stream>>>((const float4*)agg, (const float4*)b1, (v4h*)h16a, N);

    // ---- conv layer 2 ----
    gemm_wmma_kernel<0><<<blkGemm, TB, 0, stream>>>(h16a, wt2, nullptr, t32, nullptr, N);
    agg_init_kernel<<<blkN4, TB, 0, stream>>>((const float4*)t32, dinv, (float4*)agg, N);
    agg_edges_kernel<<<blkEdgeWave, TB, 0, stream>>>(src, dst, dinv, t32, agg, E);
    finalize_relu_kernel<<<blkN4, TB, 0, stream>>>((const float4*)agg, (const float4*)b2, (v4h*)h16a, N);

    // ---- readout MLP ----
    gemm_wmma_kernel<1><<<blkGemm, TB, 0, stream>>>(h16a, wtR0, Rb0, nullptr, h16b, N);
    gemm_wmma_kernel<1><<<blkGemm, TB, 0, stream>>>(h16b, wtR1, Rb1, nullptr, h16a, N);
    final_dot_kernel<<<blkN, TB, 0, stream>>>(h16a, Rw2, Rb2, out, N);
    (void)out_size; (void)n_in;
}
